// blockNL_704374637354
// MI455X (gfx1250) — matrix operand order repaired and checked by reading, not compiled, not run on hardware
//
#include <hip/hip_runtime.h>
#include <hip/hip_bf16.h>
#include <math.h>

// ---------------------------------------------------------------------------
// Problem constants
// ---------------------------------------------------------------------------
#define HIMG 512
#define WIMG 512
#define NB   4
#define CCH  31
#define CP   32          // channels padded to 32 for WMMA
#define TW   16          // tile width  (interior)
#define TH   8           // tile height (interior)
#define NPIX (TW * TH)   // 128 interior pixels per tile

// pass1: halo of 1 (one dwconv3 after the 1x1)
#define P1W  (TW + 2)          // 18
#define P1H  (TH + 2)          // 10
#define P1N  (P1W * P1H)       // 180
#define P1NP 192               // padded to multiple of 16
#define P1G  (P1NP / 16)       // 12 pixel-groups

// pass2: halo of 3 (dw3 -> dw3 -> dw3 receptive field)
#define H3W  (TW + 6)          // 22
#define H3H  (TH + 6)          // 14
#define H3N  (H3W * H3H)       // 308
#define H3NP 320
#define H3G  (H3NP / 16)       // 20 pixel-groups
#define H2W  (TW + 4)          // 20
#define H2H  (TH + 4)          // 12
#define H2N  (H2W * H2H)       // 240
#define H1W  (TW + 2)          // 18
#define H1H  (TH + 2)          // 10
#define H1N  (H1W * H1H)       // 180

typedef __attribute__((ext_vector_type(16))) __bf16 v16bf;
typedef __attribute__((ext_vector_type(8)))  float  v8f;

__device__ __forceinline__ __bf16 tobf(float f)    { return (__bf16)f; }
__device__ __forceinline__ float  frombf(__bf16 h) { return (float)h; }

// A-fragment (16x32, 16-bit) element->K mapping (contiguous order).
// k = e + 16*half
__device__ __forceinline__ int a_k(int e, int half) {
  return e + 16 * half;
}

// analytic x1: LN of a 1-channel tensor is the constant nx_b, so
// x1[d,p] = (p1_w[d]*nx_b + p1_b[d]) * S_d(p) + p2_b[d],
// with S_d(p) = sum of p2_w[d] taps whose positions fall inside the image.
__device__ __forceinline__ float x1_analytic(int d, int py, int px, float nxb,
                                             const float* __restrict__ p1_w,
                                             const float* __restrict__ p1_b,
                                             const float* __restrict__ p2_w,
                                             const float* __restrict__ p2_b) {
  float q = p1_w[d] * nxb + p1_b[d];
  float S = 0.f;
  for (int dy = -1; dy <= 1; ++dy)
    for (int dx = -1; dx <= 1; ++dx) {
      int ty = py + dy, tx = px + dx;
      if (ty >= 0 && ty < HIMG && tx >= 0 && tx < WIMG)
        S += p2_w[d * 9 + (dy + 1) * 3 + (dx + 1)];
    }
  return q * S + p2_b[d];
}

// ---------------------------------------------------------------------------
// Pass 1: LN(z) -> conv1x1(t1) [WMMA] -> dwconv3(t2) = z1 ; Gram(z1,x1) [WMMA]
// plus per-channel squared norms. Accumulated into global f32 atomics.
// ---------------------------------------------------------------------------
__global__ __launch_bounds__(256) void hsfuse_pass1(
    const float* __restrict__ z, const float* __restrict__ nz_w,
    const float* __restrict__ nz_b, const float* __restrict__ nx_b_p,
    const float* __restrict__ t1_w, const float* __restrict__ t1_b,
    const float* __restrict__ t2_w, const float* __restrict__ t2_b,
    const float* __restrict__ p1_w, const float* __restrict__ p1_b,
    const float* __restrict__ p2_w, const float* __restrict__ p2_b,
    float* __restrict__ gram, float* __restrict__ normz,
    float* __restrict__ normx) {
  __shared__ __align__(32) __bf16 sZ0[P1NP][CP];
  __shared__ __align__(32) __bf16 sT[P1NP][CP];
  __shared__ __align__(32) __bf16 sZ1T[CP][NPIX];   // z1, channel-major
  __shared__ __align__(32) __bf16 sX1T[CP][NPIX];   // x1, channel-major
  __shared__ __align__(32) __bf16 sWfrag[2][32][16]; // t1 in A-fragment layout
  __shared__ float sTB[CP];
  __shared__ float sNZ[CP];
  __shared__ float sNX[CP];

  const int tid  = threadIdx.x;
  const int lane = tid & 31;
  const int wave = tid >> 5;
  const int b    = blockIdx.z;
  const int gx0  = blockIdx.x * TW;
  const int gy0  = blockIdx.y * TH;
  const long zbase = (long)b * CCH * HIMG * WIMG;
  const float nxb = nx_b_p[0];

  // ---- stage 0: weights into LDS (A-fragment swizzle, zero padded)
  for (int idx = tid; idx < 2 * 32 * 16; idx += 256) {
    int mt = idx >> 9, ln = (idx >> 4) & 31, e = idx & 15;
    int m = (ln & 15) + 16 * mt;
    int k = a_k(e, ln >> 4);
    sWfrag[mt][ln][e] = tobf((m < CCH && k < CCH) ? t1_w[m * CCH + k] : 0.f);
  }
  if (tid < CP) {
    sTB[tid] = (tid < CCH) ? t1_b[tid] : 0.f;
    sNZ[tid] = 0.f;
    sNX[tid] = 0.f;
  }

  // ---- stage 1: LayerNorm(z) into sZ0 (bf16), zero outside image / pad rows
  for (int idx = tid; idx < P1NP; idx += 256) {
    bool ok = false;
    int py = 0, px = 0;
    if (idx < P1N) {
      py = gy0 - 1 + idx / P1W;
      px = gx0 - 1 + idx % P1W;
      ok = (py >= 0 && py < HIMG && px >= 0 && px < WIMG);
    }
    if (ok) {
      const float* zp = z + zbase + (long)py * WIMG + px;
      float v[CCH], mu = 0.f;
      for (int c = 0; c < CCH; ++c) { v[c] = zp[(long)c * HIMG * WIMG]; mu += v[c]; }
      mu *= (1.0f / CCH);
      float var = 0.f;
      for (int c = 0; c < CCH; ++c) { float d = v[c] - mu; var += d * d; }
      var *= (1.0f / CCH);
      float rs = rsqrtf(var + 1e-5f);
      for (int c = 0; c < CCH; ++c)
        sZ0[idx][c] = tobf((v[c] - mu) * rs * nz_w[c] + nz_b[c]);
      sZ0[idx][31] = tobf(0.f);
    } else {
      for (int c = 0; c < CP; ++c) sZ0[idx][c] = tobf(0.f);
    }
  }
  __syncthreads();

  // ---- stage 2: conv1x1(t1) via WMMA -> sT (all halo pixels)
  for (int t = wave; t < P1G * 2; t += 8) {
    const int g = t >> 1, mt = t & 1;
    const int half = lane >> 4;
    const int pix  = g * 16 + (lane & 15);
    v16bf B = *(const v16bf*)&sZ0[pix][16 * half];
    v16bf A = *(const v16bf*)&sWfrag[mt][lane][0];
    v8f acc;
    for (int r = 0; r < 8; ++r) acc[r] = sTB[r + 8 * half + 16 * mt];
    acc = __builtin_amdgcn_wmma_f32_16x16x32_bf16(false, A, false, B, (short)0,
                                                  acc, false, false);
    for (int r = 0; r < 8; ++r) {
      int M = r + 8 * half + 16 * mt;
      sT[pix][M] = tobf(acc[r]);
    }
  }
  __syncthreads();

  // ---- stage 2.5: dwconv3 pads with zeros outside the image
  for (int idx = tid; idx < P1N; idx += 256) {
    int py = gy0 - 1 + idx / P1W, px = gx0 - 1 + idx % P1W;
    if (py < 0 || py >= HIMG || px < 0 || px >= WIMG)
      for (int c = 0; c < CP; ++c) sT[idx][c] = tobf(0.f);
  }
  __syncthreads();

  // ---- stage 3: z1 = dwconv3(sT, t2), channel-major into sZ1T
  for (int idx = tid; idx < CP * NPIX; idx += 256) {
    int c = idx >> 7, p = idx & (NPIX - 1);
    int yi = p / TW, xi = p % TW;
    float v = 0.f;
    if (c < CCH) {
      float s = 0.f;
      for (int dy = 0; dy < 3; ++dy)
        for (int dx = 0; dx < 3; ++dx)
          s += frombf(sT[(yi + dy) * P1W + (xi + dx)][c]) * t2_w[c * 9 + dy * 3 + dx];
      v = s + t2_b[c];
    }
    sZ1T[c][p] = tobf(v);
  }

  // ---- stage 4: x1 analytic, channel-major into sX1T
  for (int idx = tid; idx < CP * NPIX; idx += 256) {
    int d = idx >> 7, p = idx & (NPIX - 1);
    float v = 0.f;
    if (d < CCH) {
      int py = gy0 + p / TW, px = gx0 + p % TW;
      v = x1_analytic(d, py, px, nxb, p1_w, p1_b, p2_w, p2_b);
    }
    sX1T[d][p] = tobf(v);
  }
  __syncthreads();

  // ---- stage 4.5: segmented norm reduction (bf16-consistent with the Gram)
  {
    int arr = tid >> 7;         // 0 -> z1 , 1 -> x1
    int c   = (tid >> 2) & 31;  // channel
    int seg = tid & 3;          // 4 x 32-pixel segments
    const __bf16* src = arr ? &sX1T[c][0] : &sZ1T[c][0];
    float s = 0.f;
    for (int p = seg * 32; p < seg * 32 + 32; ++p) {
      float v = frombf(src[p]);
      s += v * v;
    }
    atomicAdd(arr ? &sNX[c] : &sNZ[c], s);
  }

  // ---- stage 5: Gram[c,d] += sum_p z1[c,p]*x1[d,p] via WMMA
  // 4 (mt,nt) combos, 4 chunks of 32 pixels; wave w does combo w%4,
  // chunks {0,1} (w<4) or {2,3} (w>=4); accumulates then atomic-adds.
  {
    const int combo = wave & 3, mt = combo >> 1, nt = combo & 1;
    const int c0 = (wave >> 2) * 2;
    const int half = lane >> 4;
    const int m = (lane & 15) + 16 * mt;  // z channel
    const int d = (lane & 15) + 16 * nt;  // x channel
    v8f acc;
    for (int r = 0; r < 8; ++r) acc[r] = 0.f;
    for (int ch = c0; ch < c0 + 2; ++ch) {
      v16bf A, B;
      const __bf16* zrow = &sZ1T[m][ch * 32];
      for (int e = 0; e < 8; ++e) A[e]     = zrow[e + 16 * half];      // 16B run
      for (int e = 0; e < 8; ++e) A[8 + e] = zrow[8 + e + 16 * half];  // 16B run
      B = *(const v16bf*)&sX1T[d][ch * 32 + 16 * half];                // 32B run
      acc = __builtin_amdgcn_wmma_f32_16x16x32_bf16(false, A, false, B, (short)0,
                                                    acc, false, false);
    }
    for (int r = 0; r < 8; ++r) {
      int M  = r + 8 * half + 16 * mt;
      int Nc = (lane & 15) + 16 * nt;
      atomicAdd(&gram[(long)b * CP * CP + M * CP + Nc], acc[r]);
    }
  }
  __syncthreads();
  if (tid < CP) {
    atomicAdd(&normz[b * CP + tid], sNZ[tid]);
    atomicAdd(&normx[b * CP + tid], sNX[tid]);
  }
}

// ---------------------------------------------------------------------------
// Per-batch: att = softmax(gram / (|z1||x1|)); fold P = V1 * w_w * att (32x32)
// ---------------------------------------------------------------------------
__global__ __launch_bounds__(256) void hsfuse_att(
    const float* __restrict__ gram, const float* __restrict__ normz,
    const float* __restrict__ normx, const float* __restrict__ w_w,
    const float* __restrict__ v_w, float* __restrict__ Pmat) {
  __shared__ float sAtt[CP][CP];
  __shared__ float sM1[CP][CP];
  const int b = blockIdx.x, tid = threadIdx.x;

  for (int c = tid; c < CCH; c += 256) {
    float nz = fmaxf(sqrtf(fmaxf(normz[b * CP + c], 0.f)), 1e-12f);
    float row[CCH];
    float mx = -1e30f;
    for (int d = 0; d < CCH; ++d) {
      float nx = fmaxf(sqrtf(fmaxf(normx[b * CP + d], 0.f)), 1e-12f);
      float l = gram[(long)b * CP * CP + c * CP + d] / (nz * nx);
      row[d] = l;
      mx = fmaxf(mx, l);
    }
    float s = 0.f;
    for (int d = 0; d < CCH; ++d) { row[d] = expf(row[d] - mx); s += row[d]; }
    float inv = 1.f / s;
    for (int d = 0; d < CCH; ++d) sAtt[c][d] = row[d] * inv;
  }
  __syncthreads();
  // M1 = w_w @ att  (31x31)
  for (int idx = tid; idx < CCH * CCH; idx += 256) {
    int i = idx / CCH, d = idx % CCH;
    float s = 0.f;
    for (int c = 0; c < CCH; ++c) s += w_w[i * CCH + c] * sAtt[c][d];
    sM1[i][d] = s;
  }
  __syncthreads();
  // P = V1 @ M1  (32x31, pad col 31 with 0); V1[m][i] = v_w[m*32 + i + 1]
  for (int idx = tid; idx < CP * CP; idx += 256) {
    int m = idx >> 5, d = idx & 31;
    float s = 0.f;
    if (d < CCH)
      for (int i = 0; i < CCH; ++i) s += v_w[m * CP + i + 1] * sM1[i][d];
    Pmat[(long)b * CP * CP + m * CP + d] = s;
  }
}

// ---------------------------------------------------------------------------
// Pass 2: LN(z) -> conv1x1(g1) [WMMA] -> dw(g2)=z2 -> gelu(dw(pe1)) -> dw(pe2)
// y = v0*x + P@z2 + V1@(pos+z) + cvec  via two chained WMMAs per pixel group.
// ---------------------------------------------------------------------------
__global__ __launch_bounds__(256) void hsfuse_pass2(
    const float* __restrict__ x, const float* __restrict__ z,
    const float* __restrict__ nz_w, const float* __restrict__ nz_b,
    const float* __restrict__ g1_w, const float* __restrict__ g1_b,
    const float* __restrict__ g2_w, const float* __restrict__ g2_b,
    const float* __restrict__ w_b, const float* __restrict__ v_w,
    const float* __restrict__ v_b, const float* __restrict__ pe1_w,
    const float* __restrict__ pe2_w, const float* __restrict__ Pmat,
    float* __restrict__ y) {
  __shared__ __align__(32) __bf16 sA[H3NP][CP];    // z0 (halo3) -> z2 (halo2)
  __shared__ __align__(32) __bf16 sB[H3NP][CP];    // G (halo3)  -> gelu(dw) (halo1)
  __shared__ __align__(32) __bf16 sC[NPIX][CP];    // pos + z (interior)
  __shared__ __align__(32) __bf16 sGfrag[2][32][16];  // g1 fragments
  __shared__ __align__(32) __bf16 sPfrag[2][32][16];  // P = V1*W*att fragments
  __shared__ __align__(32) __bf16 sVfrag[2][32][16];  // V1 fragments
  __shared__ float sGB[CP];
  __shared__ float sCvec[CP];
  __shared__ float sV0[CP];

  const int tid  = threadIdx.x;
  const int lane = tid & 31;
  const int wave = tid >> 5;
  const int b    = blockIdx.z;
  const int gx0  = blockIdx.x * TW;
  const int gy0  = blockIdx.y * TH;
  const long zbase = (long)b * CCH * HIMG * WIMG;
  const long hw = (long)HIMG * WIMG;

  // ---- stage 0: all small matrices into LDS fragments (zero padded)
  for (int idx = tid; idx < 2 * 32 * 16; idx += 256) {
    int mt = idx >> 9, ln = (idx >> 4) & 31, e = idx & 15;
    int m = (ln & 15) + 16 * mt;
    int k = a_k(e, ln >> 4);
    sGfrag[mt][ln][e] = tobf((m < CCH && k < CCH) ? g1_w[m * CCH + k] : 0.f);
    sPfrag[mt][ln][e] = tobf(Pmat[(long)b * CP * CP + m * CP + k]); // col31 = 0
    sVfrag[mt][ln][e] = tobf((k < CCH) ? v_w[m * CP + k + 1] : 0.f);
  }
  if (tid < CP) {
    sGB[tid] = (tid < CCH) ? g1_b[tid] : 0.f;
    sV0[tid] = v_w[tid * CP];
    float s = v_b[tid];  // cvec = V1 @ w_b + v_b
    for (int d = 0; d < CCH; ++d) s += v_w[tid * CP + d + 1] * w_b[d];
    sCvec[tid] = s;
  }

  // ---- stage 1: LayerNorm(z) over halo3 into sA
  for (int idx = tid; idx < H3NP; idx += 256) {
    bool ok = false;
    int py = 0, px = 0;
    if (idx < H3N) {
      py = gy0 - 3 + idx / H3W;
      px = gx0 - 3 + idx % H3W;
      ok = (py >= 0 && py < HIMG && px >= 0 && px < WIMG);
    }
    if (ok) {
      const float* zp = z + zbase + (long)py * WIMG + px;
      float v[CCH], mu = 0.f;
      for (int c = 0; c < CCH; ++c) { v[c] = zp[(long)c * hw]; mu += v[c]; }
      mu *= (1.0f / CCH);
      float var = 0.f;
      for (int c = 0; c < CCH; ++c) { float d = v[c] - mu; var += d * d; }
      var *= (1.0f / CCH);
      float rs = rsqrtf(var + 1e-5f);
      for (int c = 0; c < CCH; ++c)
        sA[idx][c] = tobf((v[c] - mu) * rs * nz_w[c] + nz_b[c]);
      sA[idx][31] = tobf(0.f);
    } else {
      for (int c = 0; c < CP; ++c) sA[idx][c] = tobf(0.f);
    }
  }
  __syncthreads();

  // ---- stage 2: G = conv1x1(z0, g1) + g1_b via WMMA over halo3 pixels
  for (int t = wave; t < H3G * 2; t += 8) {
    const int g = t >> 1, mt = t & 1;
    const int half = lane >> 4;
    const int pix  = g * 16 + (lane & 15);
    v16bf B = *(const v16bf*)&sA[pix][16 * half];
    v16bf A = *(const v16bf*)&sGfrag[mt][lane][0];
    v8f acc;
    for (int r = 0; r < 8; ++r) acc[r] = sGB[r + 8 * half + 16 * mt];
    acc = __builtin_amdgcn_wmma_f32_16x16x32_bf16(false, A, false, B, (short)0,
                                                  acc, false, false);
    for (int r = 0; r < 8; ++r) {
      int M = r + 8 * half + 16 * mt;
      sB[pix][M] = tobf(acc[r]);
    }
  }
  __syncthreads();

  // dwconv padding: zero G outside image
  for (int idx = tid; idx < H3N; idx += 256) {
    int py = gy0 - 3 + idx / H3W, px = gx0 - 3 + idx % H3W;
    if (py < 0 || py >= HIMG || px < 0 || px >= WIMG)
      for (int c = 0; c < CP; ++c) sB[idx][c] = tobf(0.f);
  }
  __syncthreads();

  // ---- stage 3: z2 = dwconv3(G, g2) + g2_b on halo2, into sA (z0 dead)
  for (int idx = tid; idx < H2N * CP; idx += 256) {
    int p = idx >> 5, c = idx & 31;
    int y2 = p / H2W, x2 = p % H2W;
    int py = gy0 - 2 + y2, px = gx0 - 2 + x2;
    float v = 0.f;
    if (c < CCH && py >= 0 && py < HIMG && px >= 0 && px < WIMG) {
      float s = 0.f;
      for (int dy = 0; dy < 3; ++dy)
        for (int dx = 0; dx < 3; ++dx)
          s += frombf(sB[(y2 + dy) * H3W + (x2 + dx)][c]) * g2_w[c * 9 + dy * 3 + dx];
      v = s + g2_b[c];
    }
    sA[p][c] = tobf(v);
  }
  __syncthreads();

  // ---- stage 4: PG = gelu(dwconv3(z2, pe1)) on halo1, into sB (G dead)
  for (int idx = tid; idx < H1N * CP; idx += 256) {
    int p = idx >> 5, c = idx & 31;
    int y1 = p / H1W, x1c = p % H1W;
    int py = gy0 - 1 + y1, px = gx0 - 1 + x1c;
    float v = 0.f;
    if (c < CCH && py >= 0 && py < HIMG && px >= 0 && px < WIMG) {
      float s = 0.f;
      for (int dy = 0; dy < 3; ++dy)
        for (int dx = 0; dx < 3; ++dx)
          s += frombf(sA[(y1 + dy) * H2W + (x1c + dx)][c]) * pe1_w[c * 9 + dy * 3 + dx];
      v = 0.5f * s * (1.f + erff(s * 0.70710678118654752f));  // exact gelu
    }
    sB[p][c] = tobf(v);
  }
  __syncthreads();

  // ---- stage 5: pos = dwconv3(PG, pe2); pz = pos + z (interior) -> sC
  for (int idx = tid; idx < NPIX * CP; idx += 256) {
    int p = idx >> 5, c = idx & 31;
    int yi = p / TW, xi = p % TW;
    float v = 0.f;
    if (c < CCH) {
      float s = 0.f;
      for (int dy = 0; dy < 3; ++dy)
        for (int dx = 0; dx < 3; ++dx)
          s += frombf(sB[(yi + dy) * H1W + (xi + dx)][c]) * pe2_w[c * 9 + dy * 3 + dx];
      v = s + z[zbase + (long)c * hw + (long)(gy0 + yi) * WIMG + (gx0 + xi)];
    }
    sC[p][c] = tobf(v);
  }
  __syncthreads();

  // ---- stage 6: y = cvec + P@z2 + V1@(pos+z) + v0*x  (two chained WMMAs)
  {
    const int g = wave;                 // 8 groups of 16 interior pixels
    const int half = lane >> 4, ln = lane & 15;
    const int p = g * 16 + ln;
    const int yi = p / TW, xi = p % TW;
    const int py = gy0 + yi, px = gx0 + xi;
    const int idx2 = (yi + 2) * H2W + (xi + 2);  // halo2 coord of interior pixel
    v16bf B1 = *(const v16bf*)&sA[idx2][16 * half];  // z2
    v16bf B2 = *(const v16bf*)&sC[p][16 * half];     // pos + z
    const float xv = x[(long)b * hw + (long)py * WIMG + px];
    for (int mt = 0; mt < 2; ++mt) {
      v16bf Ap = *(const v16bf*)&sPfrag[mt][lane][0];
      v16bf Aq = *(const v16bf*)&sVfrag[mt][lane][0];
      v8f acc;
      for (int r = 0; r < 8; ++r) acc[r] = sCvec[r + 8 * half + 16 * mt];
      acc = __builtin_amdgcn_wmma_f32_16x16x32_bf16(false, Ap, false, B1,
                                                    (short)0, acc, false, false);
      acc = __builtin_amdgcn_wmma_f32_16x16x32_bf16(false, Aq, false, B2,
                                                    (short)0, acc, false, false);
      for (int r = 0; r < 8; ++r) {
        int M = r + 8 * half + 16 * mt;
        y[(((long)b * CP + M) * HIMG + py) * WIMG + px] = acc[r] + sV0[M] * xv;
      }
    }
  }
}

// ---------------------------------------------------------------------------
// Launch
// ---------------------------------------------------------------------------
extern "C" void kernel_launch(void* const* d_in, const int* in_sizes, int n_in,
                              void* d_out, int out_size, void* d_ws, size_t ws_size,
                              hipStream_t stream) {
  const float* x    = (const float*)d_in[0];
  const float* z    = (const float*)d_in[1];
  const float* nx_b = (const float*)d_in[3];
  const float* nz_w = (const float*)d_in[4];
  const float* nz_b = (const float*)d_in[5];
  const float* t1_w = (const float*)d_in[6];
  const float* t1_b = (const float*)d_in[7];
  const float* t2_w = (const float*)d_in[8];
  const float* t2_b = (const float*)d_in[9];
  const float* p1_w = (const float*)d_in[10];
  const float* p1_b = (const float*)d_in[11];
  const float* p2_w = (const float*)d_in[12];
  const float* p2_b = (const float*)d_in[13];
  const float* g1_w = (const float*)d_in[14];
  const float* g1_b = (const float*)d_in[15];
  const float* g2_w = (const float*)d_in[16];
  const float* g2_b = (const float*)d_in[17];
  const float* w_w  = (const float*)d_in[18];
  const float* w_b  = (const float*)d_in[19];
  const float* v_w  = (const float*)d_in[20];
  const float* v_b  = (const float*)d_in[21];
  const float* pe1_w = (const float*)d_in[22];
  const float* pe2_w = (const float*)d_in[23];
  float* y = (float*)d_out;

  float* ws    = (float*)d_ws;
  float* gram  = ws;             // NB*32*32 = 4096 floats
  float* normz = ws + 4096;      // NB*32
  float* normx = ws + 4224;      // NB*32
  float* Pmat  = ws + 4352;      // NB*32*32

  // zero the accumulators (gram + norms); Pmat fully overwritten by hsfuse_att
  hipMemsetAsync(d_ws, 0, 4352 * sizeof(float), stream);

  dim3 grid(WIMG / TW, HIMG / TH, NB);
  hsfuse_pass1<<<grid, 256, 0, stream>>>(z, nz_w, nz_b, nx_b, t1_w, t1_b, t2_w,
                                         t2_b, p1_w, p1_b, p2_w, p2_b, gram,
                                         normz, normx);
  hsfuse_att<<<NB, 256, 0, stream>>>(gram, normz, normx, w_w, v_w, Pmat);
  hsfuse_pass2<<<grid, 256, 0, stream>>>(x, z, nz_w, nz_b, g1_w, g1_b, g2_w,
                                         g2_b, w_b, v_w, v_b, pe1_w, pe2_w,
                                         Pmat, y);
}
